// Phrasal_Lexeme_42906723287547
// MI455X (gfx1250) — matrix operand order repaired and checked by reading, not compiled, not run on hardware
//
#include <hip/hip_runtime.h>
#include <hip/hip_bf16.h>

typedef __attribute__((ext_vector_type(16))) __bf16 v16bf;
typedef __attribute__((ext_vector_type(8)))  float  v8f;
typedef __attribute__((ext_vector_type(4)))  int    v4i;

typedef __attribute__((address_space(1))) v4i* gptr_v4i;   // global int4*
typedef __attribute__((address_space(3))) v4i* lptr_v4i;   // LDS int4*

constexpr int B  = 8, S = 4096, H = 1024;
constexpr int TM = 128;          // sequence rows per workgroup
constexpr int CH = H / 8;        // 16-byte (8 x bf16) chunks per row = 128
constexpr int NOT = H / 16;      // number of 16-wide o-tiles = 64

#if defined(__gfx1250__) && \
    __has_builtin(__builtin_amdgcn_global_load_async_to_lds_b128) && \
    __has_builtin(__builtin_amdgcn_s_wait_asynccnt)
#define ASYNC_LDS 1
#else
#define ASYNC_LDS 0
#endif

__device__ __forceinline__ unsigned short f2bf(float x) {
  unsigned u = __float_as_uint(x);
  u += 0x7FFFu + ((u >> 16) & 1u);          // round-to-nearest-even
  return (unsigned short)(u >> 16);
}
__device__ __forceinline__ unsigned pack2(float a, float b) {
  return (unsigned)f2bf(a) | ((unsigned)f2bf(b) << 16);
}

// ---------------- Kernel 0: W f32 -> bf16 (2 MB, stays L2-resident) --------
__global__ void convert_w_kernel(const float* __restrict__ W,
                                 uint4* __restrict__ Wb4) {
  int g = blockIdx.x * blockDim.x + threadIdx.x;   // one 16B bf16 chunk each
  const float4* W4 = (const float4*)W;
  float4 a = W4[2 * g + 0];
  float4 b = W4[2 * g + 1];
  uint4 o;
  o.x = pack2(a.x, a.y); o.y = pack2(a.z, a.w);
  o.z = pack2(b.x, b.y); o.w = pack2(b.z, b.w);
  Wb4[g] = o;
}

// ---------------- Kernel 1: fused GEMM + neighbor dot products -------------
struct SMem {
  uint4 q[TM * CH];        // 128 rows x 1024 bf16, XOR-16 swizzle (256 KB)
  uint4 w[2][16 * CH];     // double-buffered 16-wide W panels (2 x 32 KB)
};

// Stage one 16-row x H bf16 W panel into LDS (swizzled). 8 x 16B per thread.
__device__ __forceinline__ void stage_w_panel(const uint4* __restrict__ Wb4,
                                              uint4* dst, int oBase, int tid) {
  int n = tid >> 4, cb = (tid & 15) * 8;
  const uint4* src  = Wb4 + (size_t)(oBase + n) * CH;
  uint4*       drow = dst + n * CH;
#pragma unroll
  for (int cc = 0; cc < 8; ++cc) {
    int c = cb + cc;
#if ASYNC_LDS
    __builtin_amdgcn_global_load_async_to_lds_b128(
        (gptr_v4i)(src + c), (lptr_v4i)(drow + (c ^ n)), 0, 0);
#else
    drow[c ^ n] = src[c];
    __builtin_prefetch(src + CH * 16 + c, 0, 0);   // next panel hint
#endif
  }
}

__global__ void __launch_bounds__(256)
gemm_pair_dot_kernel(const float* __restrict__ query,
                     const float* __restrict__ key,
                     const uint4* __restrict__ Wb4,
                     float* __restrict__ RL,
                     float* __restrict__ RR) {
  __shared__ SMem sm;
  const int tid  = threadIdx.x;
  const int lane = tid & 31;
  const int wv   = tid >> 5;                 // 8 waves, one 16-row band each
  const int bIdx = blockIdx.x >> 5;          // 32 blocks per batch (S/TM)
  const int sWG  = (blockIdx.x & 31) << 7;

  // ---- stage query tile once: f32 -> bf16 -> LDS (swizzled) ----
  const float4* q4 = (const float4*)(query + ((size_t)bIdx * S + sWG) * H);
  for (int g = tid; g < TM * CH; g += 256) {
    int row = g >> 7, c = g & (CH - 1);
    float4 a = q4[(size_t)row * (2 * CH) + 2 * c + 0];
    float4 b = q4[(size_t)row * (2 * CH) + 2 * c + 1];
    uint4 o;
    o.x = pack2(a.x, a.y); o.y = pack2(a.z, a.w);
    o.z = pack2(b.x, b.y); o.w = pack2(b.z, b.w);
    sm.q[row * CH + (c ^ (row & 15))] = o;
  }

  // prime W panel 0 (async: lands in LDS while we finish q staging)
  stage_w_panel(Wb4, &sm.w[0][0], 0, tid);

  const int m    = lane & 15;                // A row / B column / C column
  const int hi   = lane >> 4;
  const int sRow = sWG + wv * 16;
  const uint4* qrow = &sm.q[(wv * 16 + m) * CH];
  const float* keyB = key + (size_t)bIdx * S * H;

  // clamped neighbor-row offsets (branch-free hot loop; edges zeroed later)
  unsigned offL[8], offR[8];
#pragma unroll
  for (int j = 0; j < 8; ++j) {
    int s = sRow + j + 8 * hi;
    offL[j] = (unsigned)((s > 0     ? s - 1 : 0) * H);
    offR[j] = (unsigned)((s < S - 1 ? s + 1 : 0) * H);
  }

  float accL[8] = {0.f, 0.f, 0.f, 0.f, 0.f, 0.f, 0.f, 0.f};
  float accR[8] = {0.f, 0.f, 0.f, 0.f, 0.f, 0.f, 0.f, 0.f};

  for (int ot = 0; ot < NOT; ++ot) {
    const int oBase = ot * 16;
    const int cur   = ot & 1;
#if ASYNC_LDS
    __builtin_amdgcn_s_wait_asynccnt(0);     // my panel loads have landed
#endif
    __syncthreads();                         // everyone's landed; prev reads done
    if (ot + 1 < NOT)                        // overlap next panel with compute
      stage_w_panel(Wb4, &sm.w[(ot + 1) & 1][0], oBase + 16, tid);

    v8f c8 = {};
    const uint4* wrow = &sm.w[cur][m * CH];
#pragma unroll 4
    for (int kc = 0; kc < H / 32; ++kc) {
      union { uint4 u[2]; v16bf v; } A, Bm;
      // A 16x32 bf16: lane<16 -> K{0..7,16..23}, lane>=16 -> K{8..15,24..31}
      A.u[0] = qrow[(4 * kc + hi)     ^ m];
      A.u[1] = qrow[(4 * kc + 2 + hi) ^ m];
      // B 32x16 bf16: lane<16 -> K{0..7,16..23}, lane>=16 -> K{8..15,24..31}
      int c0 = 4 * kc + hi;
      Bm.u[0] = wrow[ c0      ^ m];
      Bm.u[1] = wrow[(c0 + 2) ^ m];
      c8 = __builtin_amdgcn_wmma_f32_16x16x32_bf16(
              false, A.v, false, Bm.v, (short)0, c8, false, false);
    }

    // fused epilogue: partial dot products against neighboring key rows
    const int o = oBase + m;
#pragma unroll
    for (int j = 0; j < 8; ++j) {
      float cj = c8[j];
      float kL = keyB[offL[j] + o];
      float kR = keyB[offR[j] + o];
      accL[j] = fmaf(cj, kL, accL[j]);
      accR[j] = fmaf(cj, kR, accR[j]);
    }
  }

  // zero the clamped edge rows, then reduce over the 16 n-lanes of each half
#pragma unroll
  for (int j = 0; j < 8; ++j) {
    int s = sRow + j + 8 * hi;
    float l = (s == 0)     ? 0.f : accL[j];
    float r = (s == S - 1) ? 0.f : accR[j];
#pragma unroll
    for (int off = 8; off >= 1; off >>= 1) {
      l += __shfl_xor(l, off, 32);
      r += __shfl_xor(r, off, 32);
    }
    if (m == 0) {
      RL[(size_t)bIdx * S + s] = l;
      RR[(size_t)bIdx * S + s] = r;
    }
  }
}

// ---------------- Kernel 2: softmax pair -> sqrt -> log-sum -> exp ---------
__global__ void reduce_kernel(const float* __restrict__ RL,
                              const float* __restrict__ RR,
                              float* __restrict__ out) {
  __shared__ float red[256];
  int b = blockIdx.x, tid = threadIdx.x;
  const float* rl = RL + (size_t)b * S;
  const float* rr = RR + (size_t)b * S;
  float acc = 0.f;
  for (int i = tid; i < S - 3; i += blockDim.x) {
    int s = i + 1;
    float p1 = 1.f / (1.f + __expf(rl[s]     - rr[s]));       // pr[i,   right]
    float p0 = 1.f / (1.f + __expf(rr[s + 1] - rl[s + 1]));   // pr[i+1, left ]
    float Pi = sqrtf(p1 * p0 + 1e-9f);
    acc += __logf(Pi + 1e-8f);
  }
  red[tid] = acc;
  __syncthreads();
  for (int off = 128; off > 0; off >>= 1) {
    if (tid < off) red[tid] += red[tid + off];
    __syncthreads();
  }
  if (tid == 0) out[b] = __expf(red[0]);
}

// ---------------------------------------------------------------------------
extern "C" void kernel_launch(void* const* d_in, const int* in_sizes, int n_in,
                              void* d_out, int out_size, void* d_ws, size_t ws_size,
                              hipStream_t stream) {
  const float* query = (const float*)d_in[0];
  const float* key   = (const float*)d_in[1];
  const float* W     = (const float*)d_in[2];
  float* out = (float*)d_out;

  uint4* Wb4 = (uint4*)d_ws;                                   // 2 MB bf16 W
  float* RL  = (float*)((char*)d_ws + (size_t)H * H * 2);      // B*S f32
  float* RR  = RL + (size_t)B * S;                             // B*S f32

  convert_w_kernel<<<(H * H / 8) / 256, 256, 0, stream>>>(W, Wb4);
  gemm_pair_dot_kernel<<<(B * S) / TM, 256, 0, stream>>>(query, key, Wb4, RL, RR);
  reduce_kernel<<<B, 256, 0, stream>>>(RL, RR, out);
}
